// DQN_28784870818380
// MI455X (gfx1250) — matrix operand order repaired and checked by reading, not compiled, not run on hardware
//
#include <hip/hip_runtime.h>
#include <hip/hip_bf16.h>
#include <math.h>
#include <stdint.h>

#define NN 50000
#define NE 800000
#define NB 64
#define NA 10

typedef __attribute__((ext_vector_type(16))) __bf16 v16bf;
typedef __attribute__((ext_vector_type(8)))  __bf16 v8bf;
typedef __attribute__((ext_vector_type(8)))  float  v8f;

__device__ __forceinline__ float lrelu(float v) { return v > 0.f ? v : 0.2f * v; }

// float atomic max via signed/unsigned int ordering trick (init must be -inf)
__device__ __forceinline__ void atomicMaxF(float* addr, float v) {
    if (v >= 0.f) atomicMax((int*)addr, __float_as_int(v));
    else          atomicMin((unsigned int*)addr, (unsigned int)__float_as_int(v));
}

__global__ void k_fill(float* __restrict__ p, float v, int n) {
    int i = blockIdx.x * blockDim.x + threadIdx.x;
    if (i < n) p[i] = v;
}

// ---------------- Layer 1 node linear (K=3, scalar is right) ----------------
__global__ void k_node_lin1(const float* __restrict__ x,
                            const float* __restrict__ Wl, const float* __restrict__ bl,
                            const float* __restrict__ Wr, const float* __restrict__ br,
                            float* __restrict__ xl1, float* __restrict__ xr1) {
    int idx = blockIdx.x * blockDim.x + threadIdx.x;
    if (idx >= NN * 64) return;
    int n = idx >> 6, c = idx & 63;
    float x0 = x[n * 3 + 0], x1 = x[n * 3 + 1], x2 = x[n * 3 + 2];
    xl1[idx] = fmaf(x0, Wl[c], fmaf(x1, Wl[64 + c], fmaf(x2, Wl[128 + c], bl[c])));
    xr1[idx] = fmaf(x0, Wr[c], fmaf(x1, Wr[64 + c], fmaf(x2, Wr[128 + c], br[c])));
}

// ---------------- Edge attention logits, layer 1 (H=2, C=32) ----------------
__global__ void k_edge1_logit(const int* __restrict__ ei, const float* __restrict__ ea,
                              const float* __restrict__ xl1, const float* __restrict__ xr1,
                              const float* __restrict__ We, const float* __restrict__ att,
                              float* __restrict__ elog, float* __restrict__ dmax) {
    int e = blockIdx.x * blockDim.x + threadIdx.x;
    if (e >= NE) return;
    int src = ei[e], dst = ei[NE + e];
    if (e + 512 < NE) {   // global_prefetch_b8: hide gather latency
        __builtin_prefetch(xl1 + (size_t)ei[e + 512] * 64, 0, 1);
        __builtin_prefetch(xr1 + (size_t)ei[NE + e + 512] * 64, 0, 1);
    }
    float w = ea[e];
    const float4* sl = (const float4*)(xl1 + (size_t)src * 64);
    const float4* sr = (const float4*)(xr1 + (size_t)dst * 64);
    const float4* we = (const float4*)We;
    const float4* at = (const float4*)att;
#pragma unroll
    for (int h = 0; h < 2; h++) {
        float acc = 0.f;
#pragma unroll
        for (int q = 0; q < 8; q++) {
            float4 a = sl[h * 8 + q], b = sr[h * 8 + q];
            float4 ww = we[h * 8 + q], aa = at[h * 8 + q];
            acc += lrelu(a.x + b.x + w * ww.x) * aa.x;
            acc += lrelu(a.y + b.y + w * ww.y) * aa.y;
            acc += lrelu(a.z + b.z + w * ww.z) * aa.z;
            acc += lrelu(a.w + b.w + w * ww.w) * aa.w;
        }
        elog[e * 2 + h] = acc;
        atomicMaxF(&dmax[dst * 2 + h], acc);
    }
}

// exp(logit - segmax) + segment denominator (shared by both layers, H=2)
__global__ void k_edge_exp(const int* __restrict__ ei, float* __restrict__ elog,
                           const float* __restrict__ dmax, float* __restrict__ dsum) {
    int i = blockIdx.x * blockDim.x + threadIdx.x;
    if (i >= NE * 2) return;
    int e = i >> 1, h = i & 1;
    int dst = ei[NE + e];
    float w = expf(elog[i] - dmax[dst * 2 + h]);
    elog[i] = w;
    atomicAdd(&dsum[dst * 2 + h], w);
}

// scatter alpha * xl[src] into target accumulator, layer 1 (64 ch)
__global__ void k_edge1_scatter(const int* __restrict__ ei, const float* __restrict__ elog,
                                const float* __restrict__ dsum, const float* __restrict__ xl1,
                                float* __restrict__ hacc) {
    int idx = blockIdx.x * blockDim.x + threadIdx.x;
    if (idx >= NE * 64) return;
    int e = idx >> 6, c = idx & 63, h = c >> 5;
    int src = ei[e], dst = ei[NE + e];
    float alpha = elog[e * 2 + h] / (dsum[dst * 2 + h] + 1e-16f);
    atomicAdd(&hacc[(size_t)dst * 64 + c], alpha * xl1[(size_t)src * 64 + c]);
}

// h1 = relu(acc + bias); also produce bf16 copy for WMMA A-operand
__global__ void k_h1_final(float* __restrict__ hacc, const float* __restrict__ bias,
                           __bf16* __restrict__ h1b) {
    int idx = blockIdx.x * blockDim.x + threadIdx.x;
    if (idx >= NN * 64) return;
    float v = fmaxf(hacc[idx] + bias[idx & 63], 0.f);
    hacc[idx] = v;
    h1b[idx] = (__bf16)v;
}

// ---------------- Pre-pack W ([64 x (ntiles*16)] f32) into WMMA B-operand bf16 layout
// layout: [kt(2)][nt(ntiles)][lane(32)][elem(16)]
//   lane<16 : n = nt*16 + lane,    K = kt*32 + (elem<8 ? elem : 16 + elem-8)
//   lane>=16: n = nt*16 + lane-16, K = kt*32 + 8 + (elem<8 ? elem : 16 + elem-8)
__global__ void k_prepack_w(const float* __restrict__ W, __bf16* __restrict__ P, int ntiles) {
    int idx = blockIdx.x * blockDim.x + threadIdx.x;
    int total = 2 * ntiles * 32 * 16;
    if (idx >= total) return;
    int i    = idx & 15;
    int lane = (idx >> 4) & 31;
    int tile = idx >> 9;
    int nt   = tile % ntiles;
    int kt   = tile / ntiles;
    int ld   = ntiles << 4;
    int n  = nt * 16 + (lane & 15);
    int kk = kt * 32 + ((i >> 3) << 4) + ((lane >> 4) << 3) + (i & 7);
    P[idx] = (__bf16)W[kk * ld + n];
}

__device__ __forceinline__ v16bf load_a16(const __bf16* p0, const __bf16* p1) {
    v8bf lo = *(const v8bf*)p0;
    v8bf hi = *(const v8bf*)p1;
    v16bf r;
#pragma unroll
    for (int i = 0; i < 8; i++) { r[i] = lo[i]; r[i + 8] = hi[i]; }
    return r;
}

// ---------------- Layer 2 node transform: [50000x64] @ [64x128] x2 via WMMA --
// one wave per 16-row tile; A loaded once, reused across 8 nt x 2 matrices
__global__ void k_gemm_node2(const __bf16* __restrict__ Xb,
                             const __bf16* __restrict__ WlP, const __bf16* __restrict__ WrP,
                             const float* __restrict__ bl, const float* __restrict__ br,
                             float* __restrict__ Yl, float* __restrict__ Yr) {
    int wave = blockIdx.x * (blockDim.x >> 5) + (threadIdx.x >> 5);
    int lane = threadIdx.x & 31;
    const int mtiles = NN / 16;             // 3125 exactly
    if (wave >= mtiles) return;             // wave-uniform: EXEC stays all-ones
    int mrow  = wave * 16 + (lane & 15);
    int khalf = (lane >> 4) << 3;           // 0 or 8 (A-operand lane split)
    const __bf16* xrow = Xb + (size_t)mrow * 64;
    // A tiles: kt=0 -> K 0..31, kt=1 -> K 32..63 (16-bit A layout per ISA)
    v16bf a0 = load_a16(xrow + khalf,      xrow + 16 + khalf);
    v16bf a1 = load_a16(xrow + 32 + khalf, xrow + 48 + khalf);
    int col0  = lane & 15;
    int rbase = wave * 16 + ((lane >> 4) << 3);
#pragma unroll
    for (int nt = 0; nt < 8; nt++) {
        // lin_l
        {
            v16bf b0 = *(const v16bf*)(WlP + (size_t)((0 * 8 + nt) * 32 + lane) * 16);
            v16bf b1 = *(const v16bf*)(WlP + (size_t)((1 * 8 + nt) * 32 + lane) * 16);
            float bias = bl[nt * 16 + col0];
            v8f c;
#pragma unroll
            for (int r = 0; r < 8; r++) c[r] = bias;
            c = __builtin_amdgcn_wmma_f32_16x16x32_bf16(false, a0, false, b0, (short)0, c, false, false);
            c = __builtin_amdgcn_wmma_f32_16x16x32_bf16(false, a1, false, b1, (short)0, c, false, false);
            float* out = Yl + (size_t)rbase * 128 + nt * 16 + col0;
#pragma unroll
            for (int r = 0; r < 8; r++) out[(size_t)r * 128] = c[r];
        }
        // lin_r
        {
            v16bf b0 = *(const v16bf*)(WrP + (size_t)((0 * 8 + nt) * 32 + lane) * 16);
            v16bf b1 = *(const v16bf*)(WrP + (size_t)((1 * 8 + nt) * 32 + lane) * 16);
            float bias = br[nt * 16 + col0];
            v8f c;
#pragma unroll
            for (int r = 0; r < 8; r++) c[r] = bias;
            c = __builtin_amdgcn_wmma_f32_16x16x32_bf16(false, a0, false, b0, (short)0, c, false, false);
            c = __builtin_amdgcn_wmma_f32_16x16x32_bf16(false, a1, false, b1, (short)0, c, false, false);
            float* out = Yr + (size_t)rbase * 128 + nt * 16 + col0;
#pragma unroll
            for (int r = 0; r < 8; r++) out[(size_t)r * 128] = c[r];
        }
    }
}

// ---------------- Edge attention logits, layer 2 (H=2, C=64) ----------------
__global__ void k_edge2_logit(const int* __restrict__ ei, const float* __restrict__ ea,
                              const float* __restrict__ xl2, const float* __restrict__ xr2,
                              const float* __restrict__ We, const float* __restrict__ att,
                              float* __restrict__ elog, float* __restrict__ dmax) {
    int e = blockIdx.x * blockDim.x + threadIdx.x;
    if (e >= NE) return;
    int src = ei[e], dst = ei[NE + e];
    if (e + 512 < NE) {
        __builtin_prefetch(xl2 + (size_t)ei[e + 512] * 128, 0, 1);
        __builtin_prefetch(xr2 + (size_t)ei[NE + e + 512] * 128, 0, 1);
    }
    float w = ea[e];
    const float4* sl = (const float4*)(xl2 + (size_t)src * 128);
    const float4* sr = (const float4*)(xr2 + (size_t)dst * 128);
    const float4* we = (const float4*)We;
    const float4* at = (const float4*)att;
#pragma unroll
    for (int h = 0; h < 2; h++) {
        float acc = 0.f;
#pragma unroll
        for (int q = 0; q < 16; q++) {
            float4 a = sl[h * 16 + q], b = sr[h * 16 + q];
            float4 ww = we[h * 16 + q], aa = at[h * 16 + q];
            acc += lrelu(a.x + b.x + w * ww.x) * aa.x;
            acc += lrelu(a.y + b.y + w * ww.y) * aa.y;
            acc += lrelu(a.z + b.z + w * ww.z) * aa.z;
            acc += lrelu(a.w + b.w + w * ww.w) * aa.w;
        }
        elog[e * 2 + h] = acc;
        atomicMaxF(&dmax[dst * 2 + h], acc);
    }
}

__global__ void k_edge2_scatter(const int* __restrict__ ei, const float* __restrict__ elog,
                                const float* __restrict__ dsum, const float* __restrict__ xl2,
                                float* __restrict__ acc2) {
    int idx = blockIdx.x * blockDim.x + threadIdx.x;
    if (idx >= NE * 128) return;
    int e = idx >> 7, c = idx & 127, h = c >> 6;
    int src = ei[e], dst = ei[NE + e];
    float alpha = elog[e * 2 + h] / (dsum[dst * 2 + h] + 1e-16f);
    atomicAdd(&acc2[(size_t)dst * 128 + c], alpha * xl2[(size_t)src * 128 + c]);
}

// h2 = relu(mean over heads + bias2); also emit bf16 copy for gate WMMA
__global__ void k_h2_final(const float* __restrict__ acc2, const float* __restrict__ bias,
                           float* __restrict__ h2, __bf16* __restrict__ h2b) {
    int idx = blockIdx.x * blockDim.x + threadIdx.x;
    if (idx >= NN * 64) return;
    int n = idx >> 6, c = idx & 63;
    float v = 0.5f * (acc2[(size_t)n * 128 + c] + acc2[(size_t)n * 128 + 64 + c]) + bias[c];
    v = fmaxf(v, 0.f);
    h2[idx] = v;
    h2b[idx] = (__bf16)v;
}

// ---------------- Gate logits via WMMA: gl = h2 @ Wg + bg, fused segment max -
// [50000x64] @ [64x64]; one wave per 16-row tile, 4 column tiles
__global__ void k_gemm_gate(const __bf16* __restrict__ Xb,
                            const __bf16* __restrict__ WgP,
                            const float* __restrict__ bg, const int* __restrict__ pb,
                            float* __restrict__ gl, float* __restrict__ gmax) {
    int wave = blockIdx.x * (blockDim.x >> 5) + (threadIdx.x >> 5);
    int lane = threadIdx.x & 31;
    const int mtiles = NN / 16;
    if (wave >= mtiles) return;             // wave-uniform guard
    int mrow  = wave * 16 + (lane & 15);
    int khalf = (lane >> 4) << 3;
    const __bf16* xrow = Xb + (size_t)mrow * 64;
    v16bf a0 = load_a16(xrow + khalf,      xrow + 16 + khalf);
    v16bf a1 = load_a16(xrow + 32 + khalf, xrow + 48 + khalf);
    int col0  = lane & 15;
    int rbase = wave * 16 + ((lane >> 4) << 3);
#pragma unroll
    for (int nt = 0; nt < 4; nt++) {
        v16bf b0 = *(const v16bf*)(WgP + (size_t)((0 * 4 + nt) * 32 + lane) * 16);
        v16bf b1 = *(const v16bf*)(WgP + (size_t)((1 * 4 + nt) * 32 + lane) * 16);
        float bias = bg[nt * 16 + col0];
        v8f c;
#pragma unroll
        for (int r = 0; r < 8; r++) c[r] = bias;
        c = __builtin_amdgcn_wmma_f32_16x16x32_bf16(false, a0, false, b0, (short)0, c, false, false);
        c = __builtin_amdgcn_wmma_f32_16x16x32_bf16(false, a1, false, b1, (short)0, c, false, false);
        int col = nt * 16 + col0;
#pragma unroll
        for (int r = 0; r < 8; r++) {
            int row = rbase + r;
            gl[(size_t)row * 64 + col] = c[r];
            atomicMaxF(&gmax[pb[row] * 64 + col], c[r]);
        }
    }
}

__global__ void k_gate_exp(float* __restrict__ gl, const float* __restrict__ gmax,
                           float* __restrict__ gsum, const int* __restrict__ pb) {
    int idx = blockIdx.x * blockDim.x + threadIdx.x;
    if (idx >= NN * 64) return;
    int n = idx >> 6, c = idx & 63, b = pb[n];
    float w = expf(gl[idx] - gmax[b * 64 + c]);
    gl[idx] = w;
    atomicAdd(&gsum[b * 64 + c], w);
}

__global__ void k_gate_acc(const float* __restrict__ gl, const float* __restrict__ gsum,
                           const float* __restrict__ h2, const int* __restrict__ pb,
                           float* __restrict__ g) {
    int idx = blockIdx.x * blockDim.x + threadIdx.x;
    if (idx >= NN * 64) return;
    int n = idx >> 6, c = idx & 63, b = pb[n];
    float w = gl[idx] / (gsum[b * 64 + c] + 1e-16f);
    atomicAdd(&g[b * 64 + c], w * h2[idx]);
}

// ---------------- Generic small dense layer (B=64 rows, negligible cost) ----
__global__ void k_dense(const float* __restrict__ X, const float* __restrict__ W,
                        const float* __restrict__ bias, float* __restrict__ Y,
                        int M, int K, int Nc, int ldx, int ldy, int do_relu) {
    int idx = blockIdx.x * blockDim.x + threadIdx.x;
    if (idx >= M * Nc) return;
    int m = idx / Nc, n = idx % Nc;
    float acc = bias[n];
    const float* xr = X + (size_t)m * ldx;
    for (int k = 0; k < K; k++) acc = fmaf(xr[k], W[k * Nc + n], acc);
    if (do_relu) acc = fmaxf(acc, 0.f);
    Y[(size_t)m * ldy + n] = acc;
}

extern "C" void kernel_launch(void* const* d_in, const int* in_sizes, int n_in,
                              void* d_out, int out_size, void* d_ws, size_t ws_size,
                              hipStream_t stream) {
    (void)in_sizes; (void)n_in; (void)out_size; (void)ws_size;
    const float* x     = (const float*)d_in[0];
    const int*   ei    = (const int*)  d_in[1];
    const float* ea    = (const float*)d_in[2];
    const float* agent = (const float*)d_in[3];
    const int*   pb    = (const int*)  d_in[4];
    const float* Wl1 = (const float*)d_in[5],  *Wr1 = (const float*)d_in[6];
    const float* We1 = (const float*)d_in[7],  *at1 = (const float*)d_in[8];
    const float* Wl2 = (const float*)d_in[9],  *Wr2 = (const float*)d_in[10];
    const float* We2 = (const float*)d_in[11], *at2 = (const float*)d_in[12];
    const float* Wg  = (const float*)d_in[13];
    const float* Wf1 = (const float*)d_in[14], *Wf2 = (const float*)d_in[15];
    const float* Wa1 = (const float*)d_in[16], *Wa2 = (const float*)d_in[17];
    const float* Wa3 = (const float*)d_in[18], *Wa4 = (const float*)d_in[19];
    const float* Wo1 = (const float*)d_in[20], *Wo2 = (const float*)d_in[21];
    const float* bl1 = (const float*)d_in[22], *br1 = (const float*)d_in[23], *bias1 = (const float*)d_in[24];
    const float* bl2 = (const float*)d_in[25], *br2 = (const float*)d_in[26], *bias2 = (const float*)d_in[27];
    const float* bg  = (const float*)d_in[28];
    const float* bf1 = (const float*)d_in[29], *bf2 = (const float*)d_in[30];
    const float* ba1 = (const float*)d_in[31], *ba2 = (const float*)d_in[32];
    const float* ba3 = (const float*)d_in[33], *ba4 = (const float*)d_in[34];
    const float* bo1 = (const float*)d_in[35], *bo2 = (const float*)d_in[36];
    float* q_out = (float*)d_out;

    // linear scratch allocator (256 B aligned)
    char* ws = (char*)d_ws;
    size_t off = 0;
    auto alloc = [&](size_t bytes) -> void* {
        void* p = ws + off;
        off = (off + bytes + 255) & ~(size_t)255;
        return p;
    };
    float*  xl1   = (float*) alloc((size_t)NN * 64 * 4);
    float*  xr1   = (float*) alloc((size_t)NN * 64 * 4);
    float*  elog1 = (float*) alloc((size_t)NE * 2 * 4);
    float*  dmax1 = (float*) alloc((size_t)NN * 2 * 4);
    float*  dsum1 = (float*) alloc((size_t)NN * 2 * 4);
    float*  h1    = (float*) alloc((size_t)NN * 64 * 4);   // acc -> relu'd h1
    __bf16* h1b   = (__bf16*)alloc((size_t)NN * 64 * 2);
    __bf16* WlP   = (__bf16*)alloc((size_t)8192 * 2);
    __bf16* WrP   = (__bf16*)alloc((size_t)8192 * 2);
    __bf16* WgP   = (__bf16*)alloc((size_t)4096 * 2);
    float*  xl2   = (float*) alloc((size_t)NN * 128 * 4);
    float*  xr2   = (float*) alloc((size_t)NN * 128 * 4);
    float*  elog2 = (float*) alloc((size_t)NE * 2 * 4);
    float*  dmax2 = (float*) alloc((size_t)NN * 2 * 4);
    float*  dsum2 = (float*) alloc((size_t)NN * 2 * 4);
    float*  acc2  = (float*) alloc((size_t)NN * 128 * 4);
    float*  h2    = (float*) alloc((size_t)NN * 64 * 4);
    __bf16* h2b   = (__bf16*)alloc((size_t)NN * 64 * 2);
    float*  gl    = (float*) alloc((size_t)NN * 64 * 4);
    float*  gmax  = (float*) alloc((size_t)NB * 64 * 4);
    float*  gsum  = (float*) alloc((size_t)NB * 64 * 4);
    float*  gbuf  = (float*) alloc((size_t)NB * 64 * 4);
    float*  t1    = (float*) alloc((size_t)NB * 128 * 4);
    float*  zbuf  = (float*) alloc((size_t)NB * 96 * 4);
    float*  a1    = (float*) alloc((size_t)NB * 256 * 4);
    float*  a2    = (float*) alloc((size_t)NB * 128 * 4);
    float*  a3    = (float*) alloc((size_t)NB * 64 * 4);
    float*  o1    = (float*) alloc((size_t)NB * 128 * 4);

    auto cdiv = [](long a, long b) { return (int)((a + b - 1) / b); };
    const float NINF = -INFINITY;

    // ---- init accumulators (ws is poisoned; re-init every call) ----
    k_fill<<<cdiv((long)NN * 64, 256), 256, 0, stream>>>(h1, 0.f, NN * 64);
    k_fill<<<cdiv((long)NN * 128, 256), 256, 0, stream>>>(acc2, 0.f, NN * 128);
    k_fill<<<cdiv((long)NN * 2, 256), 256, 0, stream>>>(dmax1, NINF, NN * 2);
    k_fill<<<cdiv((long)NN * 2, 256), 256, 0, stream>>>(dsum1, 0.f, NN * 2);
    k_fill<<<cdiv((long)NN * 2, 256), 256, 0, stream>>>(dmax2, NINF, NN * 2);
    k_fill<<<cdiv((long)NN * 2, 256), 256, 0, stream>>>(dsum2, 0.f, NN * 2);
    k_fill<<<cdiv((long)NB * 64, 256), 256, 0, stream>>>(gmax, NINF, NB * 64);
    k_fill<<<cdiv((long)NB * 64, 256), 256, 0, stream>>>(gsum, 0.f, NB * 64);
    k_fill<<<cdiv((long)NB * 64, 256), 256, 0, stream>>>(gbuf, 0.f, NB * 64);

    // ---- layer 1 GATv2 ----
    k_node_lin1<<<cdiv((long)NN * 64, 256), 256, 0, stream>>>(x, Wl1, bl1, Wr1, br1, xl1, xr1);
    k_edge1_logit<<<cdiv(NE, 256), 256, 0, stream>>>(ei, ea, xl1, xr1, We1, at1, elog1, dmax1);
    k_edge_exp<<<cdiv((long)NE * 2, 256), 256, 0, stream>>>(ei, elog1, dmax1, dsum1);
    k_edge1_scatter<<<cdiv((long)NE * 64, 256), 256, 0, stream>>>(ei, elog1, dsum1, xl1, h1);
    k_h1_final<<<cdiv((long)NN * 64, 256), 256, 0, stream>>>(h1, bias1, h1b);

    // ---- layer 2 node transforms: WMMA bf16, f32 accumulate ----
    k_prepack_w<<<cdiv(8192, 256), 256, 0, stream>>>(Wl2, WlP, 8);
    k_prepack_w<<<cdiv(8192, 256), 256, 0, stream>>>(Wr2, WrP, 8);
    k_gemm_node2<<<cdiv(NN / 16, 8), 256, 0, stream>>>(h1b, WlP, WrP, bl2, br2, xl2, xr2);

    // ---- layer 2 GATv2 edge phase ----
    k_edge2_logit<<<cdiv(NE, 256), 256, 0, stream>>>(ei, ea, xl2, xr2, We2, at2, elog2, dmax2);
    k_edge_exp<<<cdiv((long)NE * 2, 256), 256, 0, stream>>>(ei, elog2, dmax2, dsum2);
    k_edge2_scatter<<<cdiv((long)NE * 128, 256), 256, 0, stream>>>(ei, elog2, dsum2, xl2, acc2);
    k_h2_final<<<cdiv((long)NN * 64, 256), 256, 0, stream>>>(acc2, bias2, h2, h2b);

    // ---- global attention pool (gate logits via WMMA + fused segment max) ----
    k_prepack_w<<<cdiv(4096, 256), 256, 0, stream>>>(Wg, WgP, 4);
    k_gemm_gate<<<cdiv(NN / 16, 8), 256, 0, stream>>>(h2b, WgP, bg, pb, gl, gmax);
    k_gate_exp<<<cdiv((long)NN * 64, 256), 256, 0, stream>>>(gl, gmax, gsum, pb);
    k_gate_acc<<<cdiv((long)NN * 64, 256), 256, 0, stream>>>(gl, gsum, h2, pb, gbuf);

    // ---- heads (B=64 rows; trivial FLOPs) ----
    // graph_fc: relu(g@Wf1+bf1)@Wf2+bf2 -> zbuf[:, 0:64]
    k_dense<<<cdiv(NB * 128, 256), 256, 0, stream>>>(gbuf, Wf1, bf1, t1, NB, 64, 128, 64, 128, 1);
    k_dense<<<cdiv(NB * 64, 256), 256, 0, stream>>>(t1, Wf2, bf2, zbuf, NB, 128, 64, 128, 96, 0);
    // agent_fc -> zbuf[:, 64:96]
    k_dense<<<cdiv(NB * 256, 256), 256, 0, stream>>>(agent, Wa1, ba1, a1, NB, 64, 256, 64, 256, 1);
    k_dense<<<cdiv(NB * 128, 256), 256, 0, stream>>>(a1, Wa2, ba2, a2, NB, 256, 128, 256, 128, 1);
    k_dense<<<cdiv(NB * 64, 256), 256, 0, stream>>>(a2, Wa3, ba3, a3, NB, 128, 64, 128, 64, 1);
    k_dense<<<cdiv(NB * 32, 256), 256, 0, stream>>>(a3, Wa4, ba4, zbuf + 64, NB, 64, 32, 64, 96, 0);
    // output head
    k_dense<<<cdiv(NB * 128, 256), 256, 0, stream>>>(zbuf, Wo1, bo1, o1, NB, 96, 128, 96, 128, 1);
    k_dense<<<cdiv(NB * NA, 256), 256, 0, stream>>>(o1, Wo2, bo2, q_out, NB, 128, NA, 128, NA, 0);
}
